// MultiHeadAttention_38448547234349
// MI455X (gfx1250) — matrix operand order repaired and checked by reading, not compiled, not run on hardware
//
#include <hip/hip_runtime.h>
#include <hip/hip_bf16.h>
#include <math.h>

typedef __attribute__((ext_vector_type(16))) __bf16 v16bf;
typedef __attribute__((ext_vector_type(8)))  float  v8f;

#define DM    1024
#define NH    16
#define DK    64
#define BATCH 64
#define SEQ   2048
#define TS    128              // S-chunk rows per block
#define NCH   (SEQ / TS)       // 16 chunks
#define NKK   (DM / 32)        // 32 K-steps of 32
#define KSTG  4                // K-steps per B staging chunk
#define NSTG  (NKK / KSTG)     // 8 staging chunks per head

// ---------------- LDS partition for the attention kernel ----------------
#define FRAG_BYTES  (8 * 32 * 32 * 32)          // A frags: 8 mtiles*32 kk*32 lanes*32B = 262144
#define BSTG_ELEMS  (4 * KSTG * 32 * 16)        // one B stage: 8192 bf16 = 16 KB
#define BSTG_BYTES  (2 * BSTG_ELEMS * 2)        // double buffered: 32 KB
#define BSTG_OFF    (FRAG_BYTES)
#define PBUF_OFF    (BSTG_OFF + BSTG_BYTES)     // 16 heads * 128 probs
#define SBUF_OFF    (PBUF_OFF + NH * TS * 4)    // 128 scores
#define RED_OFF     (SBUF_OFF + TS * 4)         // 8 floats
#define OACC_OFF    (RED_OFF + 32)              // 64 floats
#define LDS_TOTAL   (OACC_OFF + 256)            // 303,904 B <= 320 KB

// Convert weight matrix [1024x1024] fp32 (row major, k-major) into bf16
// B-fragment order: element((nt,kk,lane,i)) = W[kk*32 + (lane>=16?16:0) + i][nt*16 + (lane&15)]
__global__ __launch_bounds__(256) void wconv_kernel(const float* __restrict__ W,
                                                    __bf16* __restrict__ WB) {
    int t = blockIdx.x * 256 + threadIdx.x;
    int E = t * 4;
#pragma unroll
    for (int j = 0; j < 4; ++j) {
        int e    = E + j;
        int i    = e & 15;
        int lane = (e >> 4) & 31;
        int kk   = (e >> 9) & 31;
        int nt   = e >> 14;
        int kidx = kk * 32 + ((lane >> 4) << 4) + i;
        int n    = nt * 16 + (lane & 15);
        WB[e] = (__bf16)W[kidx * DM + n];
    }
}

// q projection: qp[b][n] = q[b]·Wq[:,n] + bq[n]   (tiny: 134 MFLOP total)
__global__ __launch_bounds__(256) void qproj_kernel(const float* __restrict__ q,
                                                    const float* __restrict__ Wq,
                                                    const float* __restrict__ bq,
                                                    float* __restrict__ qp) {
    __shared__ float qrow[DM];
    int b = blockIdx.x, t = threadIdx.x;
    for (int i = t; i < DM; i += 256) qrow[i] = q[(size_t)b * DM + i];
    __syncthreads();
#pragma unroll
    for (int j = 0; j < 4; ++j) {
        int n = t + j * 256;
        float acc = bq[n];
        for (int kidx = 0; kidx < DM; ++kidx) acc += qrow[kidx] * Wq[(size_t)kidx * DM + n];
        qp[(size_t)b * DM + n] = acc;
    }
}

// Stage one 128x1024 fp32 chunk into LDS as bf16 WMMA A-fragments.
// A-frag layout (16-bit A 16x32): lane half h selects K-halves;
// element i -> K = kk*32 + h*16 + i
__device__ __forceinline__ void load_frags(const float* __restrict__ src,
                                           __bf16* frag, int wave, int lane,
                                           int half, int lm) {
    const float* row = src + (size_t)(wave * 16 + lm) * DM;
#pragma unroll 4
    for (int kk = 0; kk < NKK; ++kk) {
        int base = kk * 32 + half * 16;
        float4 f0 = ((const float4*)(row + base))[0];
        float4 f1 = ((const float4*)(row + base))[1];
        float4 f2 = ((const float4*)(row + base + 8))[0];
        float4 f3 = ((const float4*)(row + base + 8))[1];
        v16bf a;
        a[0]=(__bf16)f0.x;  a[1]=(__bf16)f0.y;  a[2]=(__bf16)f0.z;  a[3]=(__bf16)f0.w;
        a[4]=(__bf16)f1.x;  a[5]=(__bf16)f1.y;  a[6]=(__bf16)f1.z;  a[7]=(__bf16)f1.w;
        a[8]=(__bf16)f2.x;  a[9]=(__bf16)f2.y;  a[10]=(__bf16)f2.z; a[11]=(__bf16)f2.w;
        a[12]=(__bf16)f3.x; a[13]=(__bf16)f3.y; a[14]=(__bf16)f3.z; a[15]=(__bf16)f3.w;
        *(v16bf*)(frag + ((size_t)(wave * 32 + kk) * 32 + lane) * 16) = a;
    }
}

// Fetch/store half of a 16 KB B stage (8 KB = 2 x uint4 per thread), coalesced.
__device__ __forceinline__ void bfetch2(const __bf16* __restrict__ WB, int h, int st,
                                        int tid, int half2, uint4 r[2]) {
#pragma unroll
    for (int u2 = 0; u2 < 2; ++u2) {
        int u = (half2 * 2 + u2) * 256 + tid;    // 16B unit id, 0..1023
        int nt = u >> 8, rest = u & 255;
        r[u2] = *(const uint4*)(WB +
            (((size_t)(h * 4 + nt) * 32 + st * KSTG) * 512) + (size_t)rest * 8);
    }
}
__device__ __forceinline__ void bstore2(__bf16* buf, int tid, int half2,
                                        const uint4 r[2]) {
#pragma unroll
    for (int u2 = 0; u2 < 2; ++u2) {
        int u = (half2 * 2 + u2) * 256 + tid;
        *(uint4*)(buf + (size_t)u * 8) = r[u2];
    }
}

// Two K-steps (8 WMMAs) of the per-head GEMM; B loads staggered for low pressure.
__device__ __forceinline__ void gemm_pair(const __bf16* frag, const __bf16* cur,
                                          int st, int kk0, int wave, int lane,
                                          v8f acc[4]) {
#pragma unroll
    for (int kks = kk0; kks < kk0 + 2; ++kks) {
        int kk = st * KSTG + kks;
        const __bf16* bb = cur + (size_t)kks * 512 + (size_t)lane * 16;
        v16bf a  = *(const v16bf*)(frag + ((size_t)(wave * 32 + kk) * 32 + lane) * 16);
        v16bf b0 = *(const v16bf*)(bb);
        v16bf b1 = *(const v16bf*)(bb + 1 * KSTG * 512);
        acc[0] = __builtin_amdgcn_wmma_f32_16x16x32_bf16(
            false, a, false, b0, (short)0, acc[0], false, false);
        v16bf b2 = *(const v16bf*)(bb + 2 * KSTG * 512);
        acc[1] = __builtin_amdgcn_wmma_f32_16x16x32_bf16(
            false, a, false, b1, (short)0, acc[1], false, false);
        v16bf b3 = *(const v16bf*)(bb + 3 * KSTG * 512);
        acc[2] = __builtin_amdgcn_wmma_f32_16x16x32_bf16(
            false, a, false, b2, (short)0, acc[2], false, false);
        acc[3] = __builtin_amdgcn_wmma_f32_16x16x32_bf16(
            false, a, false, b3, (short)0, acc[3], false, false);
    }
}

// [128x1024] x [1024x64] bf16 GEMM for one head; wave owns 16-row tile, 4 N-tiles.
// B is double-buffered through LDS in 16 KB stages. The next stage is fetched
// in two 8 KB halves, each drained into the (already free) other buffer in the
// middle of the current stage's compute -> 8 prefetch VGPRs peak everywhere,
// 1 barrier per stage, and each s_wait_loadcnt is covered by 8 WMMAs.
__device__ __forceinline__ void head_gemm(const __bf16* frag, __bf16* bstage,
                                          const __bf16* __restrict__ WB,
                                          int h, int wave, int lane, int tid,
                                          v8f acc[4]) {
    uint4 r[2];
    bfetch2(WB, h, 0, tid, 0, r);
    __syncthreads();                      // previous consumer of bstage done
    bstore2(bstage, tid, 0, r);
    bfetch2(WB, h, 0, tid, 1, r);         // serialized prologue: 8 regs peak
    bstore2(bstage, tid, 1, r);
    __syncthreads();
    for (int st = 0; st < NSTG; ++st) {
        const __bf16* cur = bstage + (size_t)(st & 1) * BSTG_ELEMS;
        __bf16*       nxt = bstage + (size_t)((st + 1) & 1) * BSTG_ELEMS;
        const bool pf = (st + 1 < NSTG);
        if (pf) bfetch2(WB, h, st + 1, tid, 0, r);
        gemm_pair(frag, cur, st, 0, wave, lane, acc);
        if (pf) {
            bstore2(nxt, tid, 0, r);      // nxt is free since last barrier
            bfetch2(WB, h, st + 1, tid, 1, r);
        }
        gemm_pair(frag, cur, st, 2, wave, lane, acc);
        if (pf) bstore2(nxt, tid, 1, r);
        __syncthreads();                  // nxt fully written, cur fully read
    }
}

// Fused K/V projection + flash-decoding partial attention per (batch, chunk).
// LDS-bound to 1 workgroup/WGP (exactly 2 waves per SIMD32): pin waves/EU so
// the compiler gives each wave the full half-register-file budget.
__global__ __launch_bounds__(256)
__attribute__((amdgpu_waves_per_eu(2, 2)))
void attn_kernel(
    const float* __restrict__ kin, const float* __restrict__ vin,
    const __bf16* __restrict__ WKB, const __bf16* __restrict__ WVB,
    const float* __restrict__ bk, const float* __restrict__ bv,
    const float* __restrict__ qp, float* __restrict__ ML, float* __restrict__ OP) {
    extern __shared__ char smem[];
    __bf16* frag   = (__bf16*)smem;
    __bf16* bstage = (__bf16*)(smem + BSTG_OFF);
    float*  pbuf   = (float*)(smem + PBUF_OFF);
    float*  sbuf   = (float*)(smem + SBUF_OFF);
    float*  red    = (float*)(smem + RED_OFF);
    float*  oacc   = (float*)(smem + OACC_OFF);

    const int chunk = blockIdx.x, b = blockIdx.y;
    const int tid  = threadIdx.x;
    const int wave = tid >> 5, lane = tid & 31;
    const int half = lane >> 4, lm = lane & 15;
    const size_t chunk_base = ((size_t)b * SEQ + (size_t)chunk * TS) * DM;

    // ---------- Pass 1: K chunk -> LDS frags, per-head scores + chunk softmax ----------
    load_frags(kin + chunk_base, frag, wave, lane, half, lm);
    __syncthreads();

    for (int h = 0; h < NH; ++h) {
        v8f acc[4] = {};
        head_gemm(frag, bstage, WKB, h, wave, lane, tid, acc);

        float qv[4];
#pragma unroll
        for (int nt = 0; nt < 4; ++nt) {
            int d = h * DK + nt * 16 + lm;
            qv[nt] = qp[(size_t)b * DM + d];
            float bkv = bk[d];
#pragma unroll
            for (int e = 0; e < 8; ++e) acc[nt][e] += bkv;
        }
        // scores: s[m] = (kh[m]·q_h) / 8   (reduce over n = lanes within half)
#pragma unroll
        for (int e = 0; e < 8; ++e) {
            float s = acc[0][e]*qv[0] + acc[1][e]*qv[1] + acc[2][e]*qv[2] + acc[3][e]*qv[3];
            s += __shfl_xor(s, 1, 32);
            s += __shfl_xor(s, 2, 32);
            s += __shfl_xor(s, 4, 32);
            s += __shfl_xor(s, 8, 32);
            if (lm == 0) sbuf[wave * 16 + half * 8 + e] = s * 0.125f;
        }
        __syncthreads();

        // chunk max over 128 scores
        float mv = -3.402823466e38f;
        if (tid < TS) mv = sbuf[tid];
        mv = fmaxf(mv, __shfl_xor(mv, 1, 32));
        mv = fmaxf(mv, __shfl_xor(mv, 2, 32));
        mv = fmaxf(mv, __shfl_xor(mv, 4, 32));
        mv = fmaxf(mv, __shfl_xor(mv, 8, 32));
        mv = fmaxf(mv, __shfl_xor(mv, 16, 32));
        if (tid < TS && lane == 0) red[wave] = mv;
        __syncthreads();
        float mc = fmaxf(fmaxf(red[0], red[1]), fmaxf(red[2], red[3]));

        float pv = 0.f;
        if (tid < TS) { pv = __expf(sbuf[tid] - mc); pbuf[h * TS + tid] = pv; }
        float ls = pv;
        ls += __shfl_xor(ls, 1, 32);
        ls += __shfl_xor(ls, 2, 32);
        ls += __shfl_xor(ls, 4, 32);
        ls += __shfl_xor(ls, 8, 32);
        ls += __shfl_xor(ls, 16, 32);
        __syncthreads();                      // everyone consumed red (mc)
        if (tid < TS && lane == 0) red[wave] = ls;
        __syncthreads();
        if (tid == 0) {
            float lc = red[0] + red[1] + red[2] + red[3];
            size_t ix = (((size_t)b * NH + h) * NCH + chunk) * 2;
            ML[ix] = mc; ML[ix + 1] = lc;
        }
        __syncthreads();                      // red/sbuf free for next head
    }

    // ---------- Pass 2: V chunk -> LDS frags, weighted accumulation ----------
    __syncthreads();
    load_frags(vin + chunk_base, frag, wave, lane, half, lm);
    __syncthreads();

    for (int h = 0; h < NH; ++h) {
        if (tid < DK) oacc[tid] = 0.f;
        __syncthreads();

        v8f acc[4] = {};
        head_gemm(frag, bstage, WVB, h, wave, lane, tid, acc);

#pragma unroll
        for (int nt = 0; nt < 4; ++nt) {
            float bvv = bv[h * DK + nt * 16 + lm];
            float tsum = 0.f;
#pragma unroll
            for (int e = 0; e < 8; ++e) {
                float pv = pbuf[h * TS + wave * 16 + half * 8 + e];
                tsum += pv * (acc[nt][e] + bvv);
            }
            tsum += __shfl_xor(tsum, 16, 32);  // combine the two row-halves
            if (half == 0) atomicAdd(&oacc[nt * 16 + lm], tsum);
        }
        __syncthreads();
        if (tid < DK)
            OP[(((size_t)b * NH + h) * NCH + chunk) * DK + tid] = oacc[tid];
        __syncthreads();
    }
}

// Merge chunk partials (split softmax) + output projection + ReLU.
__global__ __launch_bounds__(256) void finalize_kernel(const float* __restrict__ ML,
                                                       const float* __restrict__ OP,
                                                       const float* __restrict__ Wo,
                                                       const float* __restrict__ bo,
                                                       float* __restrict__ out) {
    __shared__ float concat[DM];
    int b = blockIdx.x, t = threadIdx.x;
    int h = t >> 4, j = t & 15;

    float M = -3.402823466e38f;
    for (int c = 0; c < NCH; ++c)
        M = fmaxf(M, ML[(((size_t)b * NH + h) * NCH + c) * 2]);
    float L = 0.f;
    for (int c = 0; c < NCH; ++c) {
        size_t ix = (((size_t)b * NH + h) * NCH + c) * 2;
        L += __expf(ML[ix] - M) * ML[ix + 1];
    }
    float invL = 1.f / L;
#pragma unroll
    for (int jj = 0; jj < 4; ++jj) {
        int d = j + jj * 16;
        float o = 0.f;
        for (int c = 0; c < NCH; ++c) {
            size_t ix = ((size_t)b * NH + h) * NCH + c;
            o += __expf(ML[ix * 2] - M) * OP[ix * DK + d];
        }
        concat[h * DK + d] = o * invL;
    }
    __syncthreads();
#pragma unroll
    for (int jj = 0; jj < 4; ++jj) {
        int n = t + jj * 256;
        float acc = bo[n];
        for (int i = 0; i < DM; ++i) acc += concat[i] * Wo[(size_t)i * DM + n];
        out[(size_t)b * DM + n] = fmaxf(acc, 0.f);
    }
}

extern "C" void kernel_launch(void* const* d_in, const int* in_sizes, int n_in,
                              void* d_out, int out_size, void* d_ws, size_t ws_size,
                              hipStream_t stream) {
    const float* q  = (const float*)d_in[0];
    const float* k  = (const float*)d_in[1];
    const float* v  = (const float*)d_in[2];
    const float* Wq = (const float*)d_in[3];
    const float* bq = (const float*)d_in[4];
    const float* Wk = (const float*)d_in[5];
    const float* bk = (const float*)d_in[6];
    const float* Wv = (const float*)d_in[7];
    const float* bv = (const float*)d_in[8];
    const float* Wo = (const float*)d_in[9];
    const float* bo = (const float*)d_in[10];
    float* out = (float*)d_out;

    char* ws = (char*)d_ws;
    __bf16* WKB = (__bf16*)ws;                                  // 2 MB
    __bf16* WVB = (__bf16*)(ws + (2u << 20));                   // 2 MB
    float*  qp  = (float*)(ws + (4u << 20));                    // 256 KB
    float*  ML  = (float*)(ws + (4u << 20) + (256u << 10));     // 128 KB
    float*  OP  = (float*)(ws + (4u << 20) + (384u << 10));     // 4 MB

    wconv_kernel<<<DM * DM / (256 * 4), 256, 0, stream>>>(Wk, WKB);
    wconv_kernel<<<DM * DM / (256 * 4), 256, 0, stream>>>(Wv, WVB);
    qproj_kernel<<<BATCH, 256, 0, stream>>>(q, Wq, bq, qp);

    dim3 grid(NCH, BATCH);
    attn_kernel<<<grid, 256, LDS_TOTAL, stream>>>(k, v, WKB, WVB, bk, bv, qp, ML, OP);

    finalize_kernel<<<BATCH, 256, 0, stream>>>(ML, OP, Wo, bo, out);
}
